// GAT_6657199309487
// MI455X (gfx1250) — compile-verified, matrix operand order checked by reading, not run on hardware
//
#include <hip/hip_runtime.h>
#include <math.h>

// GATv2 x2 + BN + mean-pool + linear for MI455X (gfx1250, wave32).
// Dense node transforms use v_wmma_f32_16x16x32_f16 (f16 in, f32 acc) with
// vectorized (b128) fragment loads: A read as two contiguous 16B runs per
// lane, B pre-swizzled into per-lane fragment order at weight-prep time.
// All four B fragments of a K-step are fetched before the first WMMA so the
// loads issue as one clause batch and waits are staged, not full barriers.
// Edge phase: two-pass segment softmax with atomics, wave-per-edge scatter.

typedef __attribute__((ext_vector_type(16))) _Float16 v16h;
typedef __attribute__((ext_vector_type(8)))  _Float16 v8h;
typedef __attribute__((ext_vector_type(8)))  float    v8f;

// ---------------- float atomic max via integer ordering trick ----------------
__device__ inline void atomicMaxF(float* addr, float val) {
  if (val >= 0.0f) {
    atomicMax((int*)addr, __float_as_int(val));
  } else {
    atomicMin((unsigned int*)addr, (unsigned int)__float_as_int(val));
  }
}

// C = A(f16) @ Bs(f16, swizzled) + bias, f32 out.
// One wave -> 16 rows x 64 cols (4 accumulator tiles).
// Bs layout: [ktile][ntile][lane 0..31][16 f16 fragment elements], so a B
// fragment is one contiguous 32B load per lane.
__global__ void gat_gemm_wmma(const _Float16* __restrict__ A,
                              const _Float16* __restrict__ Bs,
                              const float* __restrict__ bias,
                              float* __restrict__ C,
                              int M, int K, int Ncols) {
  int lane = threadIdx.x & 31;
  int wave = threadIdx.x >> 5;
  int wpb  = blockDim.x >> 5;
  int ngroup = blockIdx.y * wpb + wave;
  if (ngroup * 64 >= Ncols) return;
  int m0 = blockIdx.x * 16;
  int n0 = ngroup * 64;
  int ntiles = Ncols >> 4;
  int half = lane >> 4;

  int m = m0 + (lane & 15);
  if (m >= M) m = M - 1;
  const _Float16* rowp = A + (size_t)m * K;

  v8f acc[4];
#pragma unroll
  for (int t = 0; t < 4; ++t)
#pragma unroll
    for (int i = 0; i < 8; ++i) acc[t][i] = 0.0f;

  for (int k0 = 0, ktile = 0; k0 < K; k0 += 32, ++ktile) {
    // A fragment: two aligned 16B contiguous runs of this lane's row.
    v8h lo = *(const v8h*)(rowp + k0 + half * 8);
    v8h hi = *(const v8h*)(rowp + k0 + 16 + half * 8);

    // Fetch all four B fragments up front (distinct regs -> staged waits).
    const _Float16* bbase =
        Bs + ((((size_t)ktile * ntiles + (n0 >> 4)) * 32 + lane) << 4);
    v16h b0 = *(const v16h*)(bbase);
    v16h b1 = *(const v16h*)(bbase + (size_t)(32 << 4));
    v16h b2 = *(const v16h*)(bbase + (size_t)(2 * 32 << 4));
    v16h b3 = *(const v16h*)(bbase + (size_t)(3 * 32 << 4));

    v16h a;
#pragma unroll
    for (int i = 0; i < 8; ++i) { a[i] = lo[i]; a[8 + i] = hi[i]; }

    acc[0] = __builtin_amdgcn_wmma_f32_16x16x32_f16(
        false, a, false, b0, (short)0, acc[0], false, false);
    acc[1] = __builtin_amdgcn_wmma_f32_16x16x32_f16(
        false, a, false, b1, (short)0, acc[1], false, false);
    acc[2] = __builtin_amdgcn_wmma_f32_16x16x32_f16(
        false, a, false, b2, (short)0, acc[2], false, false);
    acc[3] = __builtin_amdgcn_wmma_f32_16x16x32_f16(
        false, a, false, b3, (short)0, acc[3], false, false);
  }

#pragma unroll
  for (int t = 0; t < 4; ++t) {
    int n = n0 + 16 * t + (lane & 15);
    float bv = bias[n];
#pragma unroll
    for (int i = 0; i < 8; ++i) {
      int row = m0 + i + 8 * half;
      if (row < M) C[(size_t)row * Ncols + n] = acc[t][i] + bv;
    }
  }
}

// ---------------- small utility kernels ----------------
__global__ void gat_fill(float* __restrict__ p, float v, long n) {
  long t = (long)blockIdx.x * blockDim.x + threadIdx.x;
  if (t < n) p[t] = v;
}

__global__ void gat_to_f16(const float* __restrict__ s, _Float16* __restrict__ d, long n) {
  long t = (long)blockIdx.x * blockDim.x + threadIdx.x;
  if (t < n) d[t] = (_Float16)s[t];
}

// Fuse Wl|Wr and swizzle into WMMA B-fragment order (f16):
// Bs[((ktile*ntiles + ntile)*32 + lane)*16 + elem]  <-  W[k][n]
// with  half = lane>>4, jj = elem>>1,
//       kb = (jj<4) ? half*8 + 2*jj : 16 + half*8 + 2*(jj-4),
//       k  = ktile*32 + kb + (elem&1),   n = ntile*16 + (lane&15).
// Also writes the fused bias vector (f32, length 2*halfN).
__global__ void gat_prep_w_swz(const float* __restrict__ Wl, const float* __restrict__ bl,
                               const float* __restrict__ Wr, const float* __restrict__ br,
                               _Float16* __restrict__ Bs, float* __restrict__ biasc,
                               int K, int halfN) {
  int Ncols = 2 * halfN;
  long total = (long)K * Ncols;
  long t = (long)blockIdx.x * blockDim.x + threadIdx.x;
  if (t < total) {
    int elem = (int)(t & 15);
    long frag = t >> 4;
    int lane = (int)(frag & 31);
    long tile = frag >> 5;
    int ntiles = Ncols >> 4;
    int ktile = (int)(tile / ntiles);
    int ntile = (int)(tile - (long)ktile * ntiles);
    int half = lane >> 4;
    int jj = elem >> 1;
    int kb = (jj < 4) ? (half * 8 + 2 * jj) : (16 + half * 8 + 2 * (jj - 4));
    int k = ktile * 32 + kb + (elem & 1);
    int n = ntile * 16 + (lane & 15);
    float w = (n < halfN) ? Wl[(size_t)k * halfN + n]
                          : Wr[(size_t)k * halfN + (n - halfN)];
    Bs[t] = (_Float16)w;
  }
  if (t < Ncols) biasc[t] = (t < halfN) ? bl[t] : br[t - halfN];
}

// ---------------- edge phase ----------------
// Pass A: alpha[e,h] = sum_c lrelu(xl[src]+xr[dst]) * att;  amax[dst,h] = max.
__global__ void gat_edge_alpha(const int* __restrict__ ei, int E, int Et, int H,
                               const float* __restrict__ xlr, int rowW, int xrOff,
                               const float* __restrict__ att,
                               float* __restrict__ alphab, float* __restrict__ amax) {
  long t = (long)blockIdx.x * blockDim.x + threadIdx.x;
  if (t >= (long)Et * H) return;
  int e = (int)(t / H);
  int h = (int)(t - (long)e * H);
  int s = (e < E) ? ei[e]     : (e - E);
  int d = (e < E) ? ei[E + e] : (e - E);
  const float4* xl = (const float4*)(xlr + (size_t)s * rowW + h * 64);
  const float4* xr = (const float4*)(xlr + (size_t)d * rowW + xrOff + h * 64);
  const float4* at = (const float4*)(att + h * 64);
  float acc = 0.0f;
#pragma unroll
  for (int j = 0; j < 16; ++j) {
    float4 a4 = xl[j], b4 = xr[j], w4 = at[j];
    float v;
    v = a4.x + b4.x; v = v > 0.0f ? v : 0.2f * v; acc += v * w4.x;
    v = a4.y + b4.y; v = v > 0.0f ? v : 0.2f * v; acc += v * w4.y;
    v = a4.z + b4.z; v = v > 0.0f ? v : 0.2f * v; acc += v * w4.z;
    v = a4.w + b4.w; v = v > 0.0f ? v : 0.2f * v; acc += v * w4.w;
  }
  alphab[t] = acc;
  atomicMaxF(&amax[(size_t)d * H + h], acc);
}

// Pass B: ex = exp(alpha - amax[dst]);  den[dst,h] += ex.
__global__ void gat_edge_expsum(const int* __restrict__ ei, int E, int Et, int H,
                                float* __restrict__ alphab,
                                const float* __restrict__ amax,
                                float* __restrict__ den) {
  long t = (long)blockIdx.x * blockDim.x + threadIdx.x;
  if (t >= (long)Et * H) return;
  int e = (int)(t / H);
  int h = (int)(t - (long)e * H);
  int d = (e < E) ? ei[E + e] : (e - E);
  float ex = __expf(alphab[t] - amax[(size_t)d * H + h]);
  alphab[t] = ex;
  atomicAdd(&den[(size_t)d * H + h], ex);
}

// Pass C: out[dst, :] += xl[src, :] * (ex / den[dst]).  One wave per edge,
// lanes sweep channels so the atomic adds are cacheline-coalesced.
__global__ void gat_edge_scatter(const int* __restrict__ ei, int E, int Et, int H,
                                 const float* __restrict__ xlr, int rowW, int outW,
                                 const float* __restrict__ alphab,
                                 const float* __restrict__ den,
                                 float* __restrict__ outacc) {
  int wpb = blockDim.x >> 5;
  int e = blockIdx.x * wpb + (threadIdx.x >> 5);
  if (e >= Et) return;
  int lane = threadIdx.x & 31;
  int s = (e < E) ? ei[e]     : (e - E);
  int d = (e < E) ? ei[E + e] : (e - E);
  const float* xl = xlr + (size_t)s * rowW;
  float* op = outacc + (size_t)d * outW;
  int iters = outW >> 5;
  for (int it = 0; it < iters; ++it) {
    int idx = lane + (it << 5);
    int h = idx >> 6;
    float a = alphab[(size_t)e * H + h] / den[(size_t)d * H + h];
    atomicAdd(&op[idx], xl[idx] * a);
  }
}

// bias + relu + BN(eval); optional f16 and/or f32 outputs (in-place allowed).
__global__ void gat_postact(const float* __restrict__ acc, const float* __restrict__ bias,
                            const float* __restrict__ g, const float* __restrict__ b,
                            const float* __restrict__ rm, const float* __restrict__ rv,
                            int cmask, long total,
                            _Float16* __restrict__ outh, float* __restrict__ outf) {
  long t = (long)blockIdx.x * blockDim.x + threadIdx.x;
  if (t >= total) return;
  int c = (int)(t & (long)cmask);
  float v = acc[t] + bias[c];
  v = fmaxf(v, 0.0f);
  v = (v - rm[c]) * (g[c] * rsqrtf(rv[c] + 1e-5f)) + b[c];
  if (outh) outh[t] = (_Float16)v;
  if (outf) outf[t] = v;
}

// mean-pool accumulation per graph
__global__ void gat_pool(const float* __restrict__ h2, const int* __restrict__ batch,
                         float* __restrict__ sums, float* __restrict__ counts, int N) {
  long t = (long)blockIdx.x * blockDim.x + threadIdx.x;
  if (t >= (long)N * 64) return;
  int n = (int)(t >> 6);
  int c = (int)(t & 63);
  int gidx = batch[n];
  atomicAdd(&sums[(size_t)gidx * 64 + c], h2[t]);
  if (c == 0) atomicAdd(&counts[gidx], 1.0f);
}

// out[g,o] = blin[o] + (1/cnt) * sum_c sums[g,c] * Wlin[c,o]
__global__ void gat_final(const float* __restrict__ sums, const float* __restrict__ counts,
                          const float* __restrict__ Wlin, const float* __restrict__ blin,
                          float* __restrict__ out, int G) {
  int t = blockIdx.x * blockDim.x + threadIdx.x;
  if (t >= G * 10) return;
  int g = t / 10;
  int o = t - g * 10;
  float inv = 1.0f / fmaxf(counts[g], 1.0f);
  const float* sg = sums + (size_t)g * 64;
  float acc = 0.0f;
#pragma unroll
  for (int c = 0; c < 64; ++c) acc += sg[c] * Wlin[c * 10 + o];
  out[t] = blin[o] + acc * inv;
}

// ---------------- host launcher ----------------
static inline int cdiv(long a, int b) { return (int)((a + b - 1) / b); }

extern "C" void kernel_launch(void* const* d_in, const int* in_sizes, int n_in,
                              void* d_out, int out_size, void* d_ws, size_t ws_size,
                              hipStream_t stream) {
  const float* x     = (const float*)d_in[0];
  const int*   ei    = (const int*)d_in[1];
  const int*   batch = (const int*)d_in[2];
  const float* Wl1   = (const float*)d_in[3];
  const float* bl1   = (const float*)d_in[4];
  const float* Wr1   = (const float*)d_in[5];
  const float* br1   = (const float*)d_in[6];
  const float* att1  = (const float*)d_in[7];
  const float* bias1 = (const float*)d_in[8];
  const float* g1    = (const float*)d_in[9];
  const float* b1    = (const float*)d_in[10];
  const float* rm1   = (const float*)d_in[11];
  const float* rv1   = (const float*)d_in[12];
  const float* Wl2   = (const float*)d_in[13];
  const float* bl2   = (const float*)d_in[14];
  const float* Wr2   = (const float*)d_in[15];
  const float* br2   = (const float*)d_in[16];
  const float* att2  = (const float*)d_in[17];
  const float* bias2 = (const float*)d_in[18];
  const float* g2    = (const float*)d_in[19];
  const float* b2    = (const float*)d_in[20];
  const float* rm2   = (const float*)d_in[21];
  const float* rv2   = (const float*)d_in[22];
  const float* Wlin  = (const float*)d_in[23];
  const float* blin  = (const float*)d_in[24];

  const int N  = in_sizes[0] / 64;
  const int E  = in_sizes[1] / 2;
  const int Et = E + N;          // with self loops
  const int G  = out_size / 10;

  // workspace carve (256B aligned)
  char* p = (char*)d_ws;
  auto carve = [&](size_t bytes) -> void* {
    void* r = (void*)p;
    p += (bytes + 255) & ~(size_t)255;
    return r;
  };
  float*    xlr1   = (float*)carve((size_t)N * 512 * 4);   // [N, 256(xl)|256(xr)]
  float*    out1   = (float*)carve((size_t)N * 256 * 4);
  float*    xlr2   = (float*)carve((size_t)N * 128 * 4);   // [N, 64(xl)|64(xr)]
  float*    out2   = (float*)carve((size_t)N * 64 * 4);
  _Float16* xh     = (_Float16*)carve((size_t)N * 64 * 2);
  _Float16* h1h    = (_Float16*)carve((size_t)N * 256 * 2);
  float*    alpha1 = (float*)carve((size_t)Et * 4 * 4);
  float*    amax1  = (float*)carve((size_t)N * 4 * 4);
  float*    den1   = (float*)carve((size_t)N * 4 * 4);
  float*    alpha2 = (float*)carve((size_t)Et * 4);
  float*    amax2  = (float*)carve((size_t)N * 4);
  float*    den2   = (float*)carve((size_t)N * 4);
  float*    sums   = (float*)carve((size_t)G * 64 * 4);
  float*    counts = (float*)carve((size_t)G * 4);
  _Float16* W1s    = (_Float16*)carve((size_t)64 * 512 * 2);   // swizzled
  float*    bias1c = (float*)carve(512 * 4);
  _Float16* W2s    = (_Float16*)carve((size_t)256 * 128 * 2);  // swizzled
  float*    bias2c = (float*)carve(128 * 4);
  (void)ws_size; (void)n_in;

  const int B = 256;
  const float NEG_INF = -__builtin_huge_valf();

  // weights -> f16, fused Wl|Wr, swizzled to B-fragment order; fused biases
  gat_prep_w_swz<<<cdiv((long)64 * 512, B), B, 0, stream>>>(Wl1, bl1, Wr1, br1, W1s, bias1c, 64, 256);
  gat_prep_w_swz<<<cdiv((long)256 * 128, B), B, 0, stream>>>(Wl2, bl2, Wr2, br2, W2s, bias2c, 256, 64);
  gat_to_f16<<<cdiv((long)N * 64, B), B, 0, stream>>>(x, xh, (long)N * 64);

  // per-call re-init (scratch poisoned / left dirty between replays)
  gat_fill<<<cdiv((long)N * 256, B), B, 0, stream>>>(out1, 0.0f, (long)N * 256);
  gat_fill<<<cdiv((long)N * 64, B), B, 0, stream>>>(out2, 0.0f, (long)N * 64);
  gat_fill<<<cdiv((long)N * 4, B), B, 0, stream>>>(amax1, NEG_INF, (long)N * 4);
  gat_fill<<<cdiv((long)N * 4, B), B, 0, stream>>>(den1, 0.0f, (long)N * 4);
  gat_fill<<<cdiv((long)N, B), B, 0, stream>>>(amax2, NEG_INF, (long)N);
  gat_fill<<<cdiv((long)N, B), B, 0, stream>>>(den2, 0.0f, (long)N);
  gat_fill<<<cdiv((long)G * 64, B), B, 0, stream>>>(sums, 0.0f, (long)G * 64);
  gat_fill<<<cdiv((long)G, B), B, 0, stream>>>(counts, 0.0f, (long)G);

  // layer 1: fused xl|xr transform via WMMA, then attention
  gat_gemm_wmma<<<dim3(cdiv(N, 16), 1, 1), 256, 0, stream>>>(xh, W1s, bias1c, xlr1, N, 64, 512);
  gat_edge_alpha<<<cdiv((long)Et * 4, B), B, 0, stream>>>(ei, E, Et, 4, xlr1, 512, 256, att1, alpha1, amax1);
  gat_edge_expsum<<<cdiv((long)Et * 4, B), B, 0, stream>>>(ei, E, Et, 4, alpha1, amax1, den1);
  gat_edge_scatter<<<cdiv(Et, B / 32), B, 0, stream>>>(ei, E, Et, 4, xlr1, 512, 256, alpha1, den1, out1);
  gat_postact<<<cdiv((long)N * 256, B), B, 0, stream>>>(out1, bias1, g1, b1, rm1, rv1, 255,
                                                        (long)N * 256, h1h, (float*)nullptr);

  // layer 2
  gat_gemm_wmma<<<dim3(cdiv(N, 16), 1, 1), 64, 0, stream>>>(h1h, W2s, bias2c, xlr2, N, 256, 128);
  gat_edge_alpha<<<cdiv((long)Et, B), B, 0, stream>>>(ei, E, Et, 1, xlr2, 128, 64, att2, alpha2, amax2);
  gat_edge_expsum<<<cdiv((long)Et, B), B, 0, stream>>>(ei, E, Et, 1, alpha2, amax2, den2);
  gat_edge_scatter<<<cdiv(Et, B / 32), B, 0, stream>>>(ei, E, Et, 1, xlr2, 128, 64, alpha2, den2, out2);
  gat_postact<<<cdiv((long)N * 64, B), B, 0, stream>>>(out2, bias2, g2, b2, rm2, rv2, 63,
                                                       (long)N * 64, (_Float16*)nullptr, out2);

  // mean pool + linear head
  gat_pool<<<cdiv((long)N * 64, B), B, 0, stream>>>(out2, batch, sums, counts, N);
  gat_final<<<cdiv((long)G * 10, B), B, 0, stream>>>(sums, counts, Wlin, blin, (float*)d_out, G);
}
